// NaturalSpeechGeneratorLoss_54760833024098
// MI455X (gfx1250) — matrix operand order repaired and checked by reading, not compiled, not run on hardware
//
#include <hip/hip_runtime.h>
#include <hip/hip_bf16.h>

// ---------------------------------------------------------------------------
// Types for CDNA5 WMMA (wave32, bf16 16x16x32 -> f32)
// ---------------------------------------------------------------------------
typedef __bf16 bf16_t;
typedef bf16_t v16bf __attribute__((ext_vector_type(16)));
typedef float  v8f   __attribute__((ext_vector_type(8)));

#define GAMMA_C   0.01f
#define INV_GAMMA 100.0f
#define WARP_C    134.4f
#define INF_C     100000.0f
#define BIG_C     1000000000.0f

// accumulator slots in workspace
#define ACC_GEN   0
#define ACC_GEN2  1
#define ACC_FM    2
#define ACC_MEL   3
#define ACC_DUR   4
#define ACC_PM    5
#define ACC_ZM    6
#define ACC_R     8   // 8..15 : soft-dtw results (dir*4 + b)

// ---------------------------------------------------------------------------
// block reduction (wave32 aware)
// ---------------------------------------------------------------------------
__device__ __forceinline__ float block_reduce_sum(float v) {
    for (int o = 16; o > 0; o >>= 1) v += __shfl_down(v, o, 32);
    __shared__ float sm[32];
    int lane = threadIdx.x & 31;
    int wid  = threadIdx.x >> 5;
    if (lane == 0) sm[wid] = v;
    __syncthreads();
    int nw = (blockDim.x + 31) >> 5;
    v = (threadIdx.x < nw) ? sm[threadIdx.x] : 0.0f;
    if (wid == 0)
        for (int o = 16; o > 0; o >>= 1) v += __shfl_down(v, o, 32);
    return v;
}

// ---------------------------------------------------------------------------
// zero accumulators
// ---------------------------------------------------------------------------
__global__ void zero_acc_kernel(float* acc) {
    if (threadIdx.x < 16) acc[threadIdx.x] = 0.0f;
}

// ---------------------------------------------------------------------------
// generic scalar reduction: op 0: sum(a); op 1: sum((1-a)^2); op 2: sum(|a-b|)
// ---------------------------------------------------------------------------
__global__ void reduce_op_kernel(const float* __restrict__ a,
                                 const float* __restrict__ b,
                                 long long n, int op, float* acc) {
    float s = 0.0f;
    long long stride = (long long)gridDim.x * blockDim.x;
    for (long long i = (long long)blockIdx.x * blockDim.x + threadIdx.x; i < n; i += stride) {
        float x = a[i];
        if (op == 0)      s += x;
        else if (op == 1) { float d = 1.0f - x; s += d * d; }
        else              s += fabsf(x - b[i]);
    }
    s = block_reduce_sum(s);
    if (threadIdx.x == 0) atomicAdd(acc, s);
}

// ---------------------------------------------------------------------------
// big bandwidth-bound |a-b| reduction (float4, ~402MB traffic -> HBM roofline)
// ---------------------------------------------------------------------------
__global__ void reduce_absdiff4_kernel(const float4* __restrict__ a,
                                       const float4* __restrict__ b,
                                       long long n4, float* acc) {
    float s = 0.0f;
    long long stride = (long long)gridDim.x * blockDim.x;
    for (long long i = (long long)blockIdx.x * blockDim.x + threadIdx.x; i < n4; i += stride) {
        __builtin_prefetch(&a[i + stride], 0, 1);   // global_prefetch_b8
        __builtin_prefetch(&b[i + stride], 0, 1);
        float4 x = a[i];
        float4 y = b[i];
        s += fabsf(x.x - y.x) + fabsf(x.y - y.y) + fabsf(x.z - y.z) + fabsf(x.w - y.w);
    }
    s = block_reduce_sum(s);
    if (threadIdx.x == 0) atomicAdd(acc, s);
}

// ---------------------------------------------------------------------------
// KL precompute, row side (i over T rows of the KL matrix):
//   A[b,i,k]   : k<H -> w = exp(-2*logs);  k>=H -> w*m         (bf16)
//   rowterm    : sum(logs) - 0.5*H + 0.5*sum(w*m^2)            (f32)
// inputs m, logs are [B,H,T] (stride T over h)
// ---------------------------------------------------------------------------
__global__ void kl_rows_kernel(const float* __restrict__ m,
                               const float* __restrict__ logs,
                               int B, int H, int T, int K,
                               bf16_t* __restrict__ Aout,
                               float* __restrict__ rowterm) {
    int tid = blockIdx.x * blockDim.x + threadIdx.x;
    if (tid >= B * T) return;
    int b = tid / T, i = tid % T;
    float slog = 0.0f, swm2 = 0.0f;
    bf16_t* arow = Aout + (size_t)tid * K;
    for (int h = 0; h < H; ++h) {
        float lp = logs[((size_t)b * H + h) * T + i];
        float mm = m   [((size_t)b * H + h) * T + i];
        float w  = __expf(-2.0f * lp);
        arow[h]     = (bf16_t)w;
        arow[H + h] = (bf16_t)(w * mm);
        slog += lp;
        swm2 += w * mm * mm;
    }
    rowterm[tid] = slog - 0.5f * (float)H + 0.5f * swm2;
}

// ---------------------------------------------------------------------------
// KL precompute, column side (j over T cols):
//   BT[b,j,k] : k<H -> z^2 ; k>=H -> -2*z    (bf16)
//   colterm   : -sum(logs_q)                  (f32)
// ---------------------------------------------------------------------------
__global__ void kl_cols_kernel(const float* __restrict__ z,
                               const float* __restrict__ logsq,
                               int B, int H, int T, int K,
                               bf16_t* __restrict__ BTout,
                               float* __restrict__ colterm) {
    int tid = blockIdx.x * blockDim.x + threadIdx.x;
    if (tid >= B * T) return;
    int b = tid / T, j = tid % T;
    float sl = 0.0f;
    bf16_t* brow = BTout + (size_t)tid * K;
    for (int h = 0; h < H; ++h) {
        float zz = z    [((size_t)b * H + h) * T + j];
        sl      += logsq[((size_t)b * H + h) * T + j];
        brow[h]     = (bf16_t)(zz * zz);
        brow[H + h] = (bf16_t)(-2.0f * zz);
    }
    colterm[tid] = -sl;
}

// ---------------------------------------------------------------------------
// WMMA GEMM: D[b,i,j] = rowterm[b,i] + colterm[b,j] + 0.5 * sum_k A[b,i,k]*BT[b,j,k]
// One wave (32 threads) per 16x16 output tile; K in steps of 32 via
// v_wmma_f32_16x16x32_bf16. Fragment layouts used here (A/B k contiguous, same order in both; not ISA table order):
//   A 16x32 bf16: lane L -> row (L&15); half=(L>>4): K in [half*16, half*16+16)
//   B 32x16 bf16: lane L -> col (L&15); half: K in [half*16, half*16+16)
//   C/D f32:      c[v] -> (M = v + 8*half, N = L&15)
// ---------------------------------------------------------------------------
__global__ void wmma_kl_kernel(const bf16_t* __restrict__ A,
                               const bf16_t* __restrict__ BT,
                               const float* __restrict__ rowterm,
                               const float* __restrict__ colterm,
                               float* __restrict__ Dout,
                               int TI, int TJ, int K) {
    int tile_j = blockIdx.x;
    int tile_i = blockIdx.y;
    int b      = blockIdx.z;

    int lane  = threadIdx.x;
    int row16 = lane & 15;
    int half  = lane >> 4;

    int ar = tile_i * 16 + row16; if (ar > TI - 1) ar = TI - 1;
    int bc = tile_j * 16 + row16; if (bc > TJ - 1) bc = TJ - 1;

    const bf16_t* arow = A  + ((size_t)b * TI + ar) * K;
    const bf16_t* brow = BT + ((size_t)b * TJ + bc) * K;

    v8f c = {};
    for (int k0 = 0; k0 < K; k0 += 32) {
        v16bf af, bfv;
        int ka = k0 + half * 16;
#pragma unroll
        for (int t = 0; t < 8; ++t) {
            af[t]     = arow[ka + t];
            af[8 + t] = arow[ka + 8 + t];
        }
        int kb = k0 + half * 16;
#pragma unroll
        for (int t = 0; t < 16; ++t) bfv[t] = brow[kb + t];

        c = __builtin_amdgcn_wmma_f32_16x16x32_bf16(
            /*neg_a=*/false, af, /*neg_b=*/false, bfv,
            /*c_mod=*/(short)0, c, /*reuse_a=*/false, /*reuse_b=*/false);
    }

    int ocol = tile_j * 16 + row16;
#pragma unroll
    for (int v = 0; v < 8; ++v) {
        int orow = tile_i * 16 + v + half * 8;
        if (orow < TI && ocol < TJ) {
            Dout[((size_t)b * TI + orow) * TJ + ocol] =
                rowterm[b * TI + orow] + colterm[b * TJ + ocol] + 0.5f * c[v];
        }
    }
}

// ---------------------------------------------------------------------------
// soft-DTW, anti-diagonal wavefront over a (TI+2) x (TJ+2) R-grid.
// One workgroup per batch; 3 rotating diagonal buffers in LDS indexed by row.
// Padded-D / mask semantics of the reference folded into the inline dval computation.
// ---------------------------------------------------------------------------
#define SDTW_MAXROWS 448

__global__ void __launch_bounds__(SDTW_MAXROWS)
softdtw_kernel(const float* __restrict__ Dm,
               const float* __restrict__ pmask,
               const float* __restrict__ qmask,
               int TI, int TJ, float* __restrict__ Rout) {
    int b = blockIdx.x;
    int t = threadIdx.x;
    int NR = TI + 2;   // rows   of R-grid (i = 0..TI+1)
    int NC = TJ + 2;   // cols   of R-grid

    __shared__ float buf[3][SDTW_MAXROWS];
    __shared__ float pmsh[SDTW_MAXROWS];
    __shared__ float qmsh[SDTW_MAXROWS];

    for (int x = t; x < TI; x += blockDim.x) pmsh[x] = pmask[(size_t)b * TI + x];
    for (int x = t; x < TJ; x += blockDim.x) qmsh[x] = qmask[(size_t)b * TJ + x];
    if (t == 0) buf[0][0] = 0.0f;   // R[0][0]
    __syncthreads();

    const float* Db = Dm + (size_t)b * TI * TJ;

    for (int d = 1; d <= NR + NC - 2; ++d) {
        int cur   = d % 3;
        int prev  = (d + 2) % 3;   // d-1
        int prev2 = (d + 1) % 3;   // d-2
        int ilo = d - (NC - 1); if (ilo < 0) ilo = 0;
        int ihi = d;            if (ihi > NR - 1) ihi = NR - 1;
        for (int i = ilo + t; i <= ihi; i += blockDim.x) {
            int j = d - i;
            float val;
            if (i == 0 || j == 0) {
                val = BIG_C;               // borders (d>=1 so never the origin)
            } else {
                float rd = buf[prev2][i - 1];
                float ru = buf[prev][i - 1];
                float rl = buf[prev][i];
                float r0 = -rd * INV_GAMMA;
                float r1 = -(ru + WARP_C) * INV_GAMMA;
                float r2 = -(rl + WARP_C) * INV_GAMMA;
                float rmax = fmaxf(r0, fmaxf(r1, r2));
                float se = __expf(r0 - rmax) + __expf(r1 - rmax) + __expf(r2 - rmax);
                float smin = -GAMMA_C * (__logf(se) + rmax);

                // padded + masked D value at padded coords (pi, pj)
                int pi = i - 1, pj = j - 1;
                bool pb = (pi < TI) && (pmsh[pi] != 0.0f);
                bool qb = (pj < TJ) && (qmsh[pj] != 0.0f);
                float dval;
                if (pb != qb)      dval = INF_C;
                else if (!pb)      dval = 0.0f;     // both masked-off / both pad
                else               dval = Db[(size_t)pi * TJ + pj];

                val = dval + smin;
                if (i == NR - 1 && j == NC - 1) Rout[0] = val;  // R[N][M]
            }
            buf[cur][i] = val;
        }
        __syncthreads();
    }
}

// ---------------------------------------------------------------------------
// final scalar combine
// ---------------------------------------------------------------------------
__global__ void finalize_kernel(const float* __restrict__ acc,
                                const float* __restrict__ dur_len,
                                float* __restrict__ out,
                                float gen_scale, float fm_scale, float mel_scale) {
    float loss_gen  = acc[ACC_GEN]  * gen_scale;
    float loss_gen2 = acc[ACC_GEN2] * gen_scale;
    float loss_fm   = acc[ACC_FM]   * fm_scale;    // includes *2*FEAT_A
    float loss_mel  = acc[ACC_MEL]  * mel_scale;   // includes *45
    float loss_dur  = acc[ACC_DUR];
    float loss_dlen = __logf(dur_len[0] + 1e-6f);
    float r1 = acc[ACC_R + 0] + acc[ACC_R + 1] + acc[ACC_R + 2] + acc[ACC_R + 3];
    float r2 = acc[ACC_R + 4] + acc[ACC_R + 5] + acc[ACC_R + 6] + acc[ACC_R + 7];
    float loss_kl     = r1 / acc[ACC_PM];
    float loss_kl_fwd = r2 / acc[ACC_ZM];
    out[0] = loss_gen + loss_gen2 + loss_fm + loss_mel + loss_dur + loss_dlen
           + loss_kl + loss_kl_fwd;
}

// ---------------------------------------------------------------------------
// host launcher
// ---------------------------------------------------------------------------
extern "C" void kernel_launch(void* const* d_in, const int* in_sizes, int n_in,
                              void* d_out, int out_size, void* d_ws, size_t ws_size,
                              hipStream_t stream) {
    const float* mel      = (const float*)d_in[0];
    const float* mel_hat  = (const float*)d_in[1];
    const float* sc       = (const float*)d_in[2];
    const float* sc_e2e   = (const float*)d_in[3];
    const float* fr       = (const float*)d_in[4];
    const float* ff       = (const float*)d_in[5];
    const float* dur      = (const float*)d_in[6];
    const float* dur_len  = (const float*)d_in[7];
    // d_in[8] = loss_pitch (unused by reference)
    const float* z_p      = (const float*)d_in[9];
    const float* m_p      = (const float*)d_in[10];
    const float* logs_p   = (const float*)d_in[11];
    const float* z_q      = (const float*)d_in[12];
    const float* m_q      = (const float*)d_in[13];
    const float* logs_q   = (const float*)d_in[14];
    const float* p_mask   = (const float*)d_in[15];
    const float* z_mask   = (const float*)d_in[16];

    const int B  = in_sizes[0] / (80 * 128);
    const int TP = in_sizes[15] / B;
    const int TQ = in_sizes[16] / B;
    const int H  = in_sizes[10] / (B * TP);
    const int K  = 2 * H;

    // ---- workspace carving (256B aligned regions) ----
    char* w = (char*)d_ws;
    auto carve = [&](size_t bytes) {
        char* p = w;
        w += (bytes + 255) & ~(size_t)255;
        return p;
    };
    float*  acc = (float*)carve(16 * sizeof(float));
    bf16_t* A1  = (bf16_t*)carve((size_t)B * TP * K * sizeof(bf16_t));
    bf16_t* B1  = (bf16_t*)carve((size_t)B * TQ * K * sizeof(bf16_t));
    bf16_t* A2  = (bf16_t*)carve((size_t)B * TQ * K * sizeof(bf16_t));
    bf16_t* B2  = (bf16_t*)carve((size_t)B * TP * K * sizeof(bf16_t));
    float*  rt1 = (float*)carve((size_t)B * TP * sizeof(float));
    float*  ct1 = (float*)carve((size_t)B * TQ * sizeof(float));
    float*  rt2 = (float*)carve((size_t)B * TQ * sizeof(float));
    float*  ct2 = (float*)carve((size_t)B * TP * sizeof(float));
    float*  D1  = (float*)carve((size_t)B * TP * TQ * sizeof(float));
    float*  D2  = (float*)carve((size_t)B * TQ * TP * sizeof(float));
    (void)ws_size; (void)n_in; (void)out_size;

    // ---- 0) zero accumulators (deterministic per call) ----
    zero_acc_kernel<<<1, 32, 0, stream>>>(acc);

    // ---- 1) cheap reductions ----
    auto grid_for = [](long long n) {
        long long g = (n + 255) / 256;
        if (g > 2048) g = 2048;
        if (g < 1)    g = 1;
        return (int)g;
    };
    long long n_sc  = in_sizes[2];
    long long n_fm  = in_sizes[4];
    long long n_mel = in_sizes[0];
    reduce_op_kernel<<<grid_for(n_sc), 256, 0, stream>>>(sc,     nullptr, n_sc, 1, acc + ACC_GEN);
    reduce_op_kernel<<<grid_for(n_sc), 256, 0, stream>>>(sc_e2e, nullptr, n_sc, 1, acc + ACC_GEN2);
    reduce_op_kernel<<<grid_for(n_mel), 256, 0, stream>>>(mel, mel_hat, n_mel, 2, acc + ACC_MEL);
    reduce_op_kernel<<<grid_for((long long)B * TP), 256, 0, stream>>>(dur,    nullptr, (long long)B * TP, 0, acc + ACC_DUR);
    reduce_op_kernel<<<grid_for((long long)B * TP), 256, 0, stream>>>(p_mask, nullptr, (long long)B * TP, 0, acc + ACC_PM);
    reduce_op_kernel<<<grid_for((long long)B * TQ), 256, 0, stream>>>(z_mask, nullptr, (long long)B * TQ, 0, acc + ACC_ZM);

    // ---- 2) the HBM-roofline reduction: sum |fr - ff| over ~50M elems ----
    long long n4   = n_fm / 4;
    long long tail = n_fm - n4 * 4;
    {
        long long g = (n4 + 255) / 256; if (g > 8192) g = 8192; if (g < 1) g = 1;
        reduce_absdiff4_kernel<<<(int)g, 256, 0, stream>>>(
            (const float4*)fr, (const float4*)ff, n4, acc + ACC_FM);
    }
    if (tail > 0)
        reduce_op_kernel<<<1, 256, 0, stream>>>(fr + n4 * 4, ff + n4 * 4, tail, 2, acc + ACC_FM);

    // ---- 3) KL operand precompute (both directions) ----
    {
        int n, g;
        n = B * TP; g = (n + 255) / 256;
        kl_rows_kernel<<<g, 256, 0, stream>>>(m_p, logs_p, B, H, TP, K, A1, rt1);   // dir1 rows (i over TP)
        kl_cols_kernel<<<g, 256, 0, stream>>>(z_q, logs_p, B, H, TP, K, B2, ct2);   // dir2 cols (j over TP)
        n = B * TQ; g = (n + 255) / 256;
        kl_cols_kernel<<<g, 256, 0, stream>>>(z_p, logs_q, B, H, TQ, K, B1, ct1);   // dir1 cols (j over TQ)
        kl_rows_kernel<<<g, 256, 0, stream>>>(m_q, logs_q, B, H, TQ, K, A2, rt2);   // dir2 rows (i over TQ)
    }

    // ---- 4) WMMA GEMMs -> KL matrices ----
    {
        dim3 g1((TQ + 15) / 16, (TP + 15) / 16, B);
        wmma_kl_kernel<<<g1, 32, 0, stream>>>(A1, B1, rt1, ct1, D1, TP, TQ, K);
        dim3 g2((TP + 15) / 16, (TQ + 15) / 16, B);
        wmma_kl_kernel<<<g2, 32, 0, stream>>>(A2, B2, rt2, ct2, D2, TQ, TP, K);
    }

    // ---- 5) soft-DTW wavefronts (B workgroups per direction; latency-bound) ----
    for (int b = 0; b < B; ++b) {
        // per-batch result pointers so the final-cell writer stores directly
        softdtw_kernel<<<1, SDTW_MAXROWS, 0, stream>>>(
            D1 + (size_t)b * TP * TQ, p_mask + (size_t)b * TP, z_mask + (size_t)b * TQ,
            TP, TQ, acc + ACC_R + 0 + b);
        softdtw_kernel<<<1, SDTW_MAXROWS, 0, stream>>>(
            D2 + (size_t)b * TQ * TP, z_mask + (size_t)b * TQ, p_mask + (size_t)b * TP,
            TQ, TP, acc + ACC_R + 4 + b);
    }

    // ---- 6) combine ----
    float gen_scale = 6.0f  / (float)n_sc;                 // sum/ (B*8192) per slice, 6 slices
    float fm_scale  = 2.0f * 24.0f / (float)n_fm;          // 2*FEAT_A * 24 / total
    float mel_scale = 45.0f / (float)n_mel;                // MEL_A * mean
    finalize_kernel<<<1, 1, 0, stream>>>(acc, dur_len, (float*)d_out,
                                         gen_scale, fm_scale, mel_scale);
}
